// AttnDecoder3_60825326846543
// MI455X (gfx1250) — matrix operand order repaired and checked by reading, not compiled, not run on hardware
//
#include <hip/hip_runtime.h>

// ---------- types ----------
typedef __attribute__((ext_vector_type(16))) __bf16 v16bf;
typedef __attribute__((ext_vector_type(8)))  __bf16 bf16x8;
typedef __attribute__((ext_vector_type(8)))  float  v8f;

// Problem constants
#define BATCH 16
#define HID   1024
#define VOC   32000
#define DECL  64
#define ENCL  256
#define TSTEPS 62   // DEC-2

// ---------- elementwise / setup kernels ----------

__global__ void k_f2bf(const float* __restrict__ s, __bf16* __restrict__ d, long n) {
  long i = (long)blockIdx.x * blockDim.x + threadIdx.x;
  if (i < n) d[i] = (__bf16)s[i];
}

__global__ void k_init(const float* __restrict__ h0, const float* __restrict__ c0,
                       float* __restrict__ h, __bf16* __restrict__ hbf, float* __restrict__ c,
                       float* __restrict__ outF, int* __restrict__ outI) {
  int i = blockIdx.x * blockDim.x + threadIdx.x;
  if (i < BATCH * HID) { float hv = h0[i]; h[i] = hv; hbf[i] = (__bf16)hv; c[i] = c0[i]; }
  if (i < BATCH * DECL) outI[i] = 0;
  if (i == 0) outF[0] = 0.f;
}

__global__ void k_biascomb(const float* __restrict__ a, const float* __restrict__ b,
                           float* __restrict__ o, int n) {
  int i = blockIdx.x * blockDim.x + threadIdx.x;
  if (i < n) o[i] = a[i] + b[i];
}

// executed[b][t] = no STOP token among idx[b][2..t+1]; time_steps = sum(executed)
__global__ void k_mask(const int* __restrict__ idx, int* __restrict__ em, float* __restrict__ ts) {
  int b = threadIdx.x;
  if (b >= BATCH) return;
  int stopped = 0, cnt = 0;
  for (int t = 0; t < TSTEPS; ++t) {
    int e = stopped ? 0 : 1;
    em[b * TSTEPS + t] = e; cnt += e;
    if (idx[b * DECL + t + 2] == 1) stopped = 1;  // STOP == 1
  }
  ts[b] = (float)cnt;
}

// ---------- WMMA GEMM:  C[M,N] = A[M,K](bf16) @ W[N,K]^T(bf16) (+bias[N]) (+addend) ----------
// Each wave computes a 16x64 strip: 4 N-tiles, one shared A fragment, 4 accumulators.
// Software double-buffering: next K-step's 5 fragments are in flight while the
// current 4 v_wmma_f32_16x16x32_bf16 execute (the register rotate carries the
// load dependency, so s_wait_loadcnt lands after the WMMA quad).
// Fragment layouts per CDNA5 ISA 7.12.2 (wave32):
//   A: lane&15 = row m; elems 0..7 at K = hi*8, elems 8..15 at K = 16+hi*8
//   B: lane&15 = col n; elems 0..7 at K = hi*8, elems 8..15 at K = 16+hi*8
//   D: acc[r] -> (m = hi*8 + r, n = lane&15)
__global__ void k_gemm(const __bf16* __restrict__ A, const __bf16* __restrict__ W,
                       const float* __restrict__ bias,
                       const float* __restrict__ add, long add_stride, long add_off,
                       float* __restrict__ C, __bf16* __restrict__ Cbf,
                       int M, int N, int K) {
  int wid   = blockIdx.x * (blockDim.x >> 5) + (threadIdx.x >> 5);
  int nquad = N >> 6;                      // N % 64 == 0 for every call site
  int tiles = (M >> 4) * nquad;
  if (wid >= tiles) return;                // whole-wave exit: EXEC stays all-1s for WMMA
  int tm = wid / nquad, tq = wid - tm * nquad;
  int lane = threadIdx.x & 31;
  int l = lane & 15, hi = lane >> 4;
  const __bf16* arow = A + (size_t)(tm * 16 + l) * K;
  const __bf16* w0   = W + (size_t)(tq * 64 + l) * K;
  const __bf16* w1   = w0 + (size_t)16 * K;
  const __bf16* w2   = w0 + (size_t)32 * K;
  const __bf16* w3   = w0 + (size_t)48 * K;

  auto ldA = [&](int k) -> v16bf {
    bf16x8 a0 = *reinterpret_cast<const bf16x8*>(arow + k + hi * 8);
    bf16x8 a1 = *reinterpret_cast<const bf16x8*>(arow + k + 16 + hi * 8);
    return __builtin_shufflevector(a0, a1, 0,1,2,3,4,5,6,7,8,9,10,11,12,13,14,15);
  };
  auto ldB = [&](const __bf16* w, int k) -> v16bf {
    bf16x8 b0 = *reinterpret_cast<const bf16x8*>(w + k + hi * 8);
    bf16x8 b1 = *reinterpret_cast<const bf16x8*>(w + k + 16 + hi * 8);
    return __builtin_shufflevector(b0, b1, 0,1,2,3,4,5,6,7,8,9,10,11,12,13,14,15);
  };

  v8f acc0 = {0.f,0.f,0.f,0.f,0.f,0.f,0.f,0.f};
  v8f acc1 = acc0, acc2 = acc0, acc3 = acc0;

  v16bf af = ldA(0);
  v16bf b0 = ldB(w0, 0), b1 = ldB(w1, 0), b2 = ldB(w2, 0), b3 = ldB(w3, 0);
  for (int k = 32; k <= K; k += 32) {
    v16bf afn = af, b0n = b0, b1n = b1, b2n = b2, b3n = b3;
    if (k < K) {                           // uniform branch; loads issue before the WMMAs retire
      afn = ldA(k);
      b0n = ldB(w0, k); b1n = ldB(w1, k); b2n = ldB(w2, k); b3n = ldB(w3, k);
    }
    acc0 = __builtin_amdgcn_wmma_f32_16x16x32_bf16(false, af, false, b0, (short)0, acc0, false, false);
    acc1 = __builtin_amdgcn_wmma_f32_16x16x32_bf16(false, af, false, b1, (short)0, acc1, false, false);
    acc2 = __builtin_amdgcn_wmma_f32_16x16x32_bf16(false, af, false, b2, (short)0, acc2, false, false);
    acc3 = __builtin_amdgcn_wmma_f32_16x16x32_bf16(false, af, false, b3, (short)0, acc3, false, false);
    af = afn; b0 = b0n; b1 = b1n; b2 = b2n; b3 = b3n;
  }

  int m0 = tm * 16 + hi * 8;
  int nbase = tq * 64 + l;
  auto emit = [&](v8f acc, int j) {
    int n = nbase + j * 16;
    float bn = bias ? bias[n] : 0.f;
#pragma unroll
    for (int r = 0; r < 8; ++r) {
      float v = acc[r] + bn;
      if (add) v += add[add_off + (size_t)(m0 + r) * add_stride + n];
      size_t o = (size_t)(m0 + r) * N + n;
      C[o] = v;
      if (Cbf) Cbf[o] = (__bf16)v;
    }
  };
  emit(acc0, 0); emit(acc1, 1); emit(acc2, 2); emit(acc3, 3);
}

// ---------- attention score:  score[b][l] = sum_f ls[b][f]*tanh(enc_proj[b][l][f]+ls[b][f]) ----------
__global__ void k_score(const float* __restrict__ encp, const float* __restrict__ ls,
                        float* __restrict__ score) {
  int bl = blockIdx.x;                // b*ENCL + l
  int b = bl >> 8;
  const float* ep = encp + (size_t)bl * (2 * HID);
  const float* lb = ls + b * (2 * HID);
  float acc = 0.f;
  for (int f = threadIdx.x; f < 2 * HID; f += 256) {
    float v = lb[f];
    acc += v * tanhf(ep[f] + v);      // v_tanh_f32 (TRANS) co-executes with VALU
  }
  __shared__ float red[256];
  red[threadIdx.x] = acc; __syncthreads();
  for (int s = 128; s > 0; s >>= 1) {
    if (threadIdx.x < s) red[threadIdx.x] += red[threadIdx.x + s];
    __syncthreads();
  }
  if (threadIdx.x == 0) score[bl] = red[0];
}

// ---------- softmax over L=256 + context vector av[b][f] = sum_l w[l]*enc[b][l][f] ----------
__global__ void k_softmax_av(const float* __restrict__ score, const float* __restrict__ enc,
                             float* __restrict__ av, __bf16* __restrict__ avbf) {
  int b = blockIdx.x, tid = threadIdx.x;
  __shared__ float w[ENCL];
  __shared__ float red[256];
  float v = score[b * ENCL + tid];
  red[tid] = v; __syncthreads();
  for (int s = 128; s > 0; s >>= 1) { if (tid < s) red[tid] = fmaxf(red[tid], red[tid + s]); __syncthreads(); }
  float mx = red[0]; __syncthreads();
  float e = __expf(v - mx);
  red[tid] = e; __syncthreads();
  for (int s = 128; s > 0; s >>= 1) { if (tid < s) red[tid] += red[tid + s]; __syncthreads(); }
  float inv = 1.f / red[0];
  w[tid] = e * inv; __syncthreads();
  const float* eb = enc + (size_t)b * ENCL * (2 * HID);
  for (int f = tid; f < 2 * HID; f += 256) {
    float a = 0.f;
    for (int l2 = 0; l2 < ENCL; ++l2) a = fmaf(w[l2], eb[(size_t)l2 * (2 * HID) + f], a);
    av[b * (2 * HID) + f] = a;
    avbf[b * (2 * HID) + f] = (__bf16)a;
  }
}

// ---------- LSTM cell pointwise (torch gate order i,f,g,o) ----------
__global__ void k_lstm(const float* __restrict__ g, float* __restrict__ c,
                       float* __restrict__ h, __bf16* __restrict__ hbf) {
  int i = blockIdx.x * blockDim.x + threadIdx.x;   // BATCH*HID
  int b = i >> 10, j = i & (HID - 1);
  const float* gb = g + b * 4 * HID;
  float gi = gb[j], gf = gb[HID + j], gc = gb[2 * HID + j], go = gb[3 * HID + j];
  float si = 1.f / (1.f + __expf(-gi));
  float sf = 1.f / (1.f + __expf(-gf));
  float so = 1.f / (1.f + __expf(-go));
  float c2 = sf * c[i] + si * tanhf(gc);
  float h2 = so * tanhf(c2);
  c[i] = c2; h[i] = h2; hbf[i] = (__bf16)h2;
}

// ---------- per-step argmax + log-softmax loss over V=32000 ----------
__global__ void k_loss(const float* __restrict__ logits, const int* __restrict__ idx,
                       const int* __restrict__ em, float* __restrict__ losses,
                       int* __restrict__ outI, int t) {
  int b = blockIdx.x, tid = threadIdx.x;
  const float* lg = logits + (size_t)b * VOC;
  __shared__ float smax[256];
  __shared__ int   sidx[256];
  float mx = -3.4e38f; int mi = 0;
  for (int j = tid; j < VOC; j += 256) { float v = lg[j]; if (v > mx) { mx = v; mi = j; } }
  smax[tid] = mx; sidx[tid] = mi; __syncthreads();
  for (int s = 128; s > 0; s >>= 1) {
    if (tid < s) {
      float vo = smax[tid], vn = smax[tid + s];
      if (vn > vo || (vn == vo && sidx[tid + s] < sidx[tid])) { smax[tid] = vn; sidx[tid] = sidx[tid + s]; }
    }
    __syncthreads();
  }
  float gmax = smax[0]; int gidx = sidx[0]; __syncthreads();
  float se = 0.f;
  for (int j = tid; j < VOC; j += 256) se += __expf(lg[j] - gmax);
  smax[tid] = se; __syncthreads();
  for (int s = 128; s > 0; s >>= 1) { if (tid < s) smax[tid] += smax[tid + s]; __syncthreads(); }
  if (tid == 0) {
    float lse = gmax + __logf(smax[0]);
    int tgt = idx[b * DECL + t + 1];
    losses[b * TSTEPS + t] = lse - lg[tgt];
    outI[b * DECL + t + 1] = em[b * TSTEPS + t] ? gidx : 0;
  }
}

// ---------- final: loss_sum = sum_b (sum_t loss*exec)/time_steps ----------
__global__ void k_lossred(const float* __restrict__ losses, const int* __restrict__ em,
                          const float* __restrict__ ts, float* __restrict__ out) {
  __shared__ float red[32];
  int b = threadIdx.x;
  float s = 0.f;
  if (b < BATCH) {
    for (int t = 0; t < TSTEPS; ++t) s += losses[b * TSTEPS + t] * (float)em[b * TSTEPS + t];
    s /= ts[b];
  }
  red[b] = s; __syncthreads();
  if (b == 0) { float tot = 0.f; for (int i = 0; i < BATCH; ++i) tot += red[i]; out[0] = tot; }
}

// ---------- host driver ----------
extern "C" void kernel_launch(void* const* d_in, const int* in_sizes, int n_in,
                              void* d_out, int out_size, void* d_ws, size_t ws_size,
                              hipStream_t stream) {
  (void)in_sizes; (void)n_in; (void)out_size; (void)ws_size;
  const float* decode_inputs = (const float*)d_in[0];
  const int*   dii           = (const int*)d_in[1];
  const float* h0   = (const float*)d_in[2];
  const float* c0   = (const float*)d_in[3];
  const float* enc  = (const float*)d_in[4];
  const float* W_conv = (const float*)d_in[5];
  const float* b_conv = (const float*)d_in[6];
  const float* W_hid  = (const float*)d_in[7];
  const float* b_hid  = (const float*)d_in[8];
  const float* W_inl  = (const float*)d_in[9];
  const float* b_inl  = (const float*)d_in[10];
  const float* W_ih   = (const float*)d_in[11];
  const float* W_hh   = (const float*)d_in[12];
  const float* b_ih   = (const float*)d_in[13];
  const float* b_hh   = (const float*)d_in[14];
  const float* W_outl = (const float*)d_in[15];
  const float* b_outl = (const float*)d_in[16];
  const float* W_out  = (const float*)d_in[17];
  const float* b_out  = (const float*)d_in[18];

  char* base = (char*)d_ws; size_t off = 0;
  auto alloc = [&](size_t bytes) -> void* {
    void* p = base + off;
    off += (bytes + 255) & ~(size_t)255;
    return p;
  };
  // bf16 copies (weights + encoder): ~114 MB -> L2-resident on MI455X (192 MB L2)
  __bf16* enc_bf   = (__bf16*)alloc((size_t)BATCH * ENCL * 2 * HID * 2);
  __bf16* Wc_bf    = (__bf16*)alloc((size_t)2 * HID * 2 * HID * 2);
  __bf16* Whid_bf  = (__bf16*)alloc((size_t)2 * HID * HID * 2);
  __bf16* Winl_bf  = (__bf16*)alloc((size_t)(HID / 2) * 2 * HID * 2);
  __bf16* Wih_bf   = (__bf16*)alloc((size_t)4 * HID * (HID / 2) * 2);
  __bf16* Whh_bf   = (__bf16*)alloc((size_t)4 * HID * HID * 2);
  __bf16* Woutl_bf = (__bf16*)alloc((size_t)HID * 2 * HID * 2);
  __bf16* Wout_bf  = (__bf16*)alloc((size_t)VOC * HID * 2);
  // fp32 activations / state
  float* enc_proj = (float*)alloc((size_t)BATCH * ENCL * 2 * HID * 4);
  float* ls       = (float*)alloc((size_t)BATCH * 2 * HID * 4);
  float* score    = (float*)alloc((size_t)BATCH * ENCL * 4);
  float*  av      = (float*)alloc((size_t)BATCH * 2 * HID * 4);
  __bf16* av_bf   = (__bf16*)alloc((size_t)BATCH * 2 * HID * 2);
  float*  inp     = (float*)alloc((size_t)BATCH * (HID / 2) * 4);
  __bf16* inp_bf  = (__bf16*)alloc((size_t)BATCH * (HID / 2) * 2);
  float*  gates   = (float*)alloc((size_t)BATCH * 4 * HID * 4);
  float*  h       = (float*)alloc((size_t)BATCH * HID * 4);
  __bf16* h_bf    = (__bf16*)alloc((size_t)BATCH * HID * 2);
  float*  c       = (float*)alloc((size_t)BATCH * HID * 4);
  float*  proj    = (float*)alloc((size_t)BATCH * HID * 4);
  __bf16* proj_bf = (__bf16*)alloc((size_t)BATCH * HID * 2);
  float*  logits  = (float*)alloc((size_t)BATCH * VOC * 4);
  float*  losses  = (float*)alloc((size_t)BATCH * TSTEPS * 4);
  int*    execm   = (int*)alloc((size_t)BATCH * TSTEPS * 4);
  float*  tsteps  = (float*)alloc((size_t)BATCH * 4);
  float*  biasc   = (float*)alloc((size_t)4 * HID * 4);

  float* outF = (float*)d_out;
  int*   outI = ((int*)d_out) + 1;

  auto cvt = [&](const float* s, __bf16* d, long n) {
    k_f2bf<<<dim3((unsigned)((n + 255) / 256)), dim3(256), 0, stream>>>(s, d, n);
  };
  auto gemm = [&](const __bf16* A, const __bf16* W, const float* bias,
                  const float* add, long astride, long aoff,
                  float* C, __bf16* Cbf, int M, int N, int K) {
    int tiles = (M / 16) * (N / 64);       // 16x64 strip per wave
    int blocks = (tiles + 3) / 4;
    k_gemm<<<dim3(blocks), dim3(128), 0, stream>>>(A, W, bias, add, astride, aoff,
                                                   C, Cbf, M, N, K);
  };

  // ---- setup (deterministic each call) ----
  cvt(enc,    enc_bf,   (long)BATCH * ENCL * 2 * HID);
  cvt(W_conv, Wc_bf,    (long)2 * HID * 2 * HID);
  cvt(W_hid,  Whid_bf,  (long)2 * HID * HID);
  cvt(W_inl,  Winl_bf,  (long)(HID / 2) * 2 * HID);
  cvt(W_ih,   Wih_bf,   (long)4 * HID * (HID / 2));
  cvt(W_hh,   Whh_bf,   (long)4 * HID * HID);
  cvt(W_outl, Woutl_bf, (long)HID * 2 * HID);
  cvt(W_out,  Wout_bf,  (long)VOC * HID);
  k_init<<<64, 256, 0, stream>>>(h0, c0, h, h_bf, c, outF, outI);
  k_mask<<<1, 32, 0, stream>>>(dii, execm, tsteps);
  k_biascomb<<<16, 256, 0, stream>>>(b_ih, b_hh, biasc, 4 * HID);

  // enc_proj = enc @ W_conv^T + b_conv  (34 GFLOP, once)
  gemm(enc_bf, Wc_bf, b_conv, nullptr, 0, 0, enc_proj, nullptr,
       BATCH * ENCL, 2 * HID, 2 * HID);

  // av0 = attend(h0)
  gemm(h_bf, Whid_bf, b_hid, nullptr, 0, 0, ls, nullptr, BATCH, 2 * HID, HID);
  k_score<<<BATCH * ENCL, 256, 0, stream>>>(enc_proj, ls, score);
  k_softmax_av<<<BATCH, 256, 0, stream>>>(score, enc, av, av_bf);

  // ---- 62 recurrent steps (stream order encodes the dependency chain) ----
  for (int t = 0; t < TSTEPS; ++t) {
    // inp = x_t + av @ W_inl^T + b_inl
    gemm(av_bf, Winl_bf, b_inl, decode_inputs, (long)DECL * (HID / 2), (long)t * (HID / 2),
         inp, inp_bf, BATCH, HID / 2, 2 * HID);
    // gates = inp @ W_ih^T + (b_ih+b_hh) + h @ W_hh^T
    gemm(inp_bf, Wih_bf, biasc, nullptr, 0, 0, gates, nullptr, BATCH, 4 * HID, HID / 2);
    gemm(h_bf, Whh_bf, nullptr, gates, 4 * HID, 0, gates, nullptr, BATCH, 4 * HID, HID);
    // (h,c) update
    k_lstm<<<64, 256, 0, stream>>>(gates, c, h, h_bf);
    // attend(h2)
    gemm(h_bf, Whid_bf, b_hid, nullptr, 0, 0, ls, nullptr, BATCH, 2 * HID, HID);
    k_score<<<BATCH * ENCL, 256, 0, stream>>>(enc_proj, ls, score);
    k_softmax_av<<<BATCH, 256, 0, stream>>>(score, enc, av, av_bf);
    // proj = av2 @ W_outl^T + b_outl + h2 ; logits = proj @ W_out^T + b_out
    gemm(av_bf, Woutl_bf, b_outl, h, HID, 0, proj, proj_bf, BATCH, HID, 2 * HID);
    gemm(proj_bf, Wout_bf, b_out, nullptr, 0, 0, logits, nullptr, BATCH, VOC, HID);
    // argmax + loss
    k_loss<<<BATCH, 256, 0, stream>>>(logits, dii, execm, losses, outI, t);
  }
  k_lossred<<<1, 32, 0, stream>>>(losses, execm, tsteps, outF);
}
